// GPT_38036230373992
// MI455X (gfx1250) — matrix operand order repaired and checked by reading, not compiled, not run on hardware
//
#include <hip/hip_runtime.h>
#include <hip/hip_bf16.h>
#include <math.h>

// ---------------------------------------------------------------------------
// GPT-2 forward (B=2, T=1024, L=12, D=768, H=12, HD=64, V=50257) for gfx1250.
// GEMMs: bf16 WMMA (v_wmma_f32_16x16x32_bf16), fp32 accumulate.
// 128x128 block tile, 8 waves (32x64 each = 8 WMMA/step), double-buffered LDS.
// ---------------------------------------------------------------------------

typedef __attribute__((ext_vector_type(16))) __bf16 v16bf;
typedef __attribute__((ext_vector_type(8)))  __bf16 v8bf;
typedef __attribute__((ext_vector_type(8)))  float  v8f;

#define TSTR 48   // LDS row stride in bf16 halves (96B: 32B-aligned fragments, depadded banks)

__device__ __forceinline__ float gelu_tanh(float x) {
    float x3 = x * x * x;
    return 0.5f * x * (1.0f + tanhf(0.7978845608028654f * (x + 0.044715f * x3)));
}

// ---- token + position embedding -------------------------------------------
__global__ void k_embed(const int* __restrict__ idx, const float* __restrict__ wte,
                        const float* __restrict__ wpe, float* __restrict__ x) {
    int row = blockIdx.x;          // b*1024 + t
    int t   = row & 1023;
    int id  = idx[row];
    const float* we = wte + (size_t)id * 768;
    const float* pe = wpe + (size_t)t * 768;
    float* xo = x + (size_t)row * 768;
    for (int j = threadIdx.x; j < 768; j += blockDim.x)
        xo[j] = we[j] + pe[j];
}

// ---- layernorm (one block per row of 768) ----------------------------------
__global__ void k_layernorm(const float* __restrict__ x, const float* __restrict__ g,
                            const float* __restrict__ b, float* __restrict__ o) {
    __shared__ float red[256];
    int row = blockIdx.x;
    const float* xr = x + (size_t)row * 768;
    float s = 0.f, s2 = 0.f;
    for (int j = threadIdx.x; j < 768; j += 256) { float v = xr[j]; s += v; s2 += v * v; }
    red[threadIdx.x] = s; __syncthreads();
    for (int st = 128; st > 0; st >>= 1) {
        if (threadIdx.x < st) red[threadIdx.x] += red[threadIdx.x + st];
        __syncthreads();
    }
    float mu = red[0] * (1.0f / 768.0f);
    __syncthreads();
    red[threadIdx.x] = s2; __syncthreads();
    for (int st = 128; st > 0; st >>= 1) {
        if (threadIdx.x < st) red[threadIdx.x] += red[threadIdx.x + st];
        __syncthreads();
    }
    float var = red[0] * (1.0f / 768.0f) - mu * mu;
    float rs  = rsqrtf(var + 1e-5f);
    float* orow = o + (size_t)row * 768;
    for (int j = threadIdx.x; j < 768; j += 256)
        orow[j] = (xr[j] - mu) * rs * g[j] + b[j];
}

// ---- WMMA GEMM: C[M,N] = act(A[M,K] @ W + bias) + res ----------------------
// blayout==0: W is [K][N] row-major (weights).  blayout==1: W is [N][K] (wte).
// 256 threads = 8 waves; 128x128 block tile; wave = 32x64 (2x4 WMMA tiles).
// Double-buffered LDS staging: stage tile kt+1 while computing tile kt.
__global__ __launch_bounds__(256)
void k_gemm(const float* __restrict__ A, const float* __restrict__ W,
            const float* __restrict__ bias, const float* __restrict__ res,
            float* __restrict__ C, int M, int N, int K, int blayout, int act) {
    __shared__ __bf16 Alds[2][128 * TSTR];   // [buf][row 0..127][k 0..31]
    __shared__ __bf16 Blds[2][128 * TSTR];   // [buf] B^T: [n 0..127][k 0..31]

    const int tid  = threadIdx.x;
    const int lane = tid & 31;
    const int wid  = tid >> 5;
    const int wm   = (wid & 3) * 32;     // wave row offset within 128-tile
    const int wn   = (wid >> 2) * 64;    // wave col offset within 128-tile
    const int lr   = lane & 15;
    const int hs   = lane >> 4;          // half-select (0/1)
    const int nBase = blockIdx.x * 128;
    const int mBase = blockIdx.y * 128;

    v8f acc[2][4] = {};

    const int nk = K >> 5;

    auto stage = [&](int buf, int kt) {
        const int k0 = kt * 32;
        // A tile: 128x32 fp32 -> bf16
        {
            int rowl = tid >> 1, cb = (tid & 1) * 16;
            const float* ag = A + (size_t)(mBase + rowl) * K + k0 + cb;
            __bf16* dst = &Alds[buf][rowl * TSTR + cb];
            #pragma unroll
            for (int i = 0; i < 16; ++i) dst[i] = (__bf16)ag[i];
        }
        // B tile transposed into Blds[n][k]
        if (blayout == 0) {
            int kr = tid >> 3, nb = (tid & 7) * 16;
            const float* bg = W + (size_t)(k0 + kr) * N + nBase + nb;
            #pragma unroll
            for (int i = 0; i < 16; ++i) {
                float v = (nBase + nb + i < N) ? bg[i] : 0.f;
                Blds[buf][(nb + i) * TSTR + kr] = (__bf16)v;
            }
        } else {
            int nr = tid >> 1, cb = (tid & 1) * 16;
            int gn = nBase + nr;
            const float* bg = W + (size_t)gn * K + k0 + cb;
            __bf16* dst = &Blds[buf][nr * TSTR + cb];
            if (gn < N) {
                #pragma unroll
                for (int i = 0; i < 16; ++i) dst[i] = (__bf16)bg[i];
            } else {
                #pragma unroll
                for (int i = 0; i < 16; ++i) dst[i] = (__bf16)0.f;
            }
        }
    };

    stage(0, 0);
    __syncthreads();

    for (int kt = 0; kt < nk; ++kt) {
        const int cur = kt & 1;
        if (kt + 1 < nk) stage(1 - cur, kt + 1);   // overlap next-tile staging

        // fragments per documented wave32 layouts:
        // A 16x32 bf16: lane row = lr, kbase = hs*8; halves [0..7]=K kbase..+7,
        //               halves [8..15]=K kbase+16..+23 -> two 16B LDS loads.
        v16bf a[2];
        #pragma unroll
        for (int ti = 0; ti < 2; ++ti) {
            const __bf16* ar = &Alds[cur][(wm + ti * 16 + lr) * TSTR + hs * 8];
            v8bf lo = *(const v8bf*)ar, hi = *(const v8bf*)(ar + 16);
            #pragma unroll
            for (int i = 0; i < 8; ++i) { a[ti][i] = lo[i]; a[ti][i + 8] = hi[i]; }
        }
        // B 32x16 bf16: lane col = lr, same K order as A -> two 16B LDS loads.
        v16bf b[4];
        #pragma unroll
        for (int tj = 0; tj < 4; ++tj) {
            const __bf16* br = &Blds[cur][(wn + tj * 16 + lr) * TSTR + hs * 8];
            v8bf lo = *(const v8bf*)br, hi = *(const v8bf*)(br + 16);
            #pragma unroll
            for (int i = 0; i < 8; ++i) { b[tj][i] = lo[i]; b[tj][i + 8] = hi[i]; }
        }

        #pragma unroll
        for (int ti = 0; ti < 2; ++ti)
            #pragma unroll
            for (int tj = 0; tj < 4; ++tj)
                acc[ti][tj] = __builtin_amdgcn_wmma_f32_16x16x32_bf16(
                    false, a[ti], false, b[tj], (short)0, acc[ti][tj], false, false);

        __syncthreads();  // publish staged buffer; protect the one just read
    }

    // --- epilogue: bias (+gelu) (+residual), store ---
    #pragma unroll
    for (int ti = 0; ti < 2; ++ti) {
        #pragma unroll
        for (int tj = 0; tj < 4; ++tj) {
            int col = nBase + wn + tj * 16 + lr;
            if (col >= N) continue;
            float badd = bias ? bias[col] : 0.f;
            v8f c = acc[ti][tj];
            #pragma unroll
            for (int r = 0; r < 8; ++r) {
                int rowG = mBase + wm + ti * 16 + hs * 8 + r;   // C: M = r + 8*hs
                float v = c[r] + badd;
                if (act) v = gelu_tanh(v);
                size_t off = (size_t)rowG * N + col;
                if (res) v += res[off];
                C[off] = v;
            }
        }
    }
}

// ---- causal attention (flash-style online softmax), 1 thread = 1 query row -
__global__ __launch_bounds__(64)
void k_attn(const float* __restrict__ qkv, float* __restrict__ y) {
    __shared__ float Kl[64 * 65];
    __shared__ float Vl[64 * 65];
    const int tid = threadIdx.x;
    const int qt = blockIdx.x, h = blockIdx.y, b = blockIdx.z;
    const int bT = b * 1024;
    const int q  = qt * 64 + tid;
    const size_t qoff = (size_t)(bT + q) * 2304 + h * 64;

    float Q[64], acc[64];
    #pragma unroll
    for (int d = 0; d < 64; ++d) { Q[d] = qkv[qoff + d]; acc[d] = 0.f; }
    float m = -3.4e38f, l = 0.f;

    for (int kb = 0; kb <= qt; ++kb) {
        __syncthreads();
        for (int r = 0; r < 64; ++r) {
            size_t ko = (size_t)(bT + kb * 64 + r) * 2304 + h * 64 + tid;
            Kl[r * 65 + tid] = qkv[ko + 768];
            Vl[r * 65 + tid] = qkv[ko + 1536];
        }
        __syncthreads();
        int kmax = q - kb * 64;            // causal: key index <= q
        if (kmax > 63) kmax = 63;
        for (int kk = 0; kk <= kmax; ++kk) {
            float s = 0.f;
            #pragma unroll
            for (int d = 0; d < 64; ++d) s += Q[d] * Kl[kk * 65 + d];
            s *= 0.125f;                   // 1/sqrt(64)
            float mn = fmaxf(m, s);
            float alpha = __expf(m - mn);
            float p = __expf(s - mn);
            l = l * alpha + p;
            #pragma unroll
            for (int d = 0; d < 64; ++d) acc[d] = acc[d] * alpha + p * Vl[kk * 65 + d];
            m = mn;
        }
    }
    float inv = 1.f / l;
    float* yo = y + (size_t)(bT + q) * 768 + h * 64;
    #pragma unroll
    for (int d = 0; d < 64; ++d) yo[d] = acc[d] * inv;
}

// ---- per-row cross-entropy: rl[row] = logsumexp(row) - logits[row][tgt] ----
__global__ void k_rowloss(const float* __restrict__ logits, const int* __restrict__ tgt,
                          float* __restrict__ rl) {
    __shared__ float red[256];
    int row = blockIdx.x;
    const float* lr = logits + (size_t)row * 50257;
    float mx = -3.4e38f;
    for (int j = threadIdx.x; j < 50257; j += 256) mx = fmaxf(mx, lr[j]);
    red[threadIdx.x] = mx; __syncthreads();
    for (int st = 128; st > 0; st >>= 1) {
        if (threadIdx.x < st) red[threadIdx.x] = fmaxf(red[threadIdx.x], red[threadIdx.x + st]);
        __syncthreads();
    }
    mx = red[0]; __syncthreads();
    float s = 0.f;
    for (int j = threadIdx.x; j < 50257; j += 256) s += __expf(lr[j] - mx);
    red[threadIdx.x] = s; __syncthreads();
    for (int st = 128; st > 0; st >>= 1) {
        if (threadIdx.x < st) red[threadIdx.x] += red[threadIdx.x + st];
        __syncthreads();
    }
    if (threadIdx.x == 0) {
        float lse = mx + logf(red[0]);
        rl[row] = lse - lr[tgt[row]];
    }
}

__global__ void k_finloss(const float* __restrict__ rl, float* __restrict__ out) {
    __shared__ float red[256];
    float s = 0.f;
    for (int j = threadIdx.x; j < 2048; j += 256) s += rl[j];
    red[threadIdx.x] = s; __syncthreads();
    for (int st = 128; st > 0; st >>= 1) {
        if (threadIdx.x < st) red[threadIdx.x] += red[threadIdx.x + st];
        __syncthreads();
    }
    if (threadIdx.x == 0) out[0] = red[0] * (1.0f / 2048.0f);
}

// ---------------------------------------------------------------------------
extern "C" void kernel_launch(void* const* d_in, const int* in_sizes, int n_in,
                              void* d_out, int out_size, void* d_ws, size_t ws_size,
                              hipStream_t stream) {
    (void)in_sizes; (void)n_in; (void)out_size; (void)ws_size;
    const int*   idx   = (const int*)d_in[0];
    const int*   tgt   = (const int*)d_in[1];
    const float* wte   = (const float*)d_in[2];
    const float* wpe   = (const float*)d_in[3];
    const float* lnf_g = (const float*)d_in[4];
    const float* lnf_b = (const float*)d_in[5];

    float* logits = (float*)d_out;                 // [2048][50257]
    float* x   = (float*)d_ws;                     // [2048][768]
    float* ln  = x   + 1572864;                    // [2048][768]
    float* qkv = ln  + 1572864;                    // [2048][2304]
    float* hb  = qkv + 4718592;                    // [2048][3072] (also attn y)
    float* rl  = hb  + 6291456;                    // [2048]

    k_embed<<<2048, 256, 0, stream>>>(idx, wte, wpe, x);

    for (int li = 0; li < 12; ++li) {
        const float* const* p = (const float* const*)(d_in + 6 + 12 * li);
        // p: 0 ln1_g, 1 ln1_b, 2 attn_w, 3 attn_b, 4 attn_proj_w, 5 attn_proj_b,
        //    6 ln2_g, 7 ln2_b, 8 fc_w, 9 fc_b, 10 fc_proj_w, 11 fc_proj_b
        k_layernorm<<<2048, 256, 0, stream>>>(x, p[0], p[1], ln);
        k_gemm<<<dim3(18, 16), 256, 0, stream>>>(ln, p[2], p[3], nullptr, qkv,
                                                 2048, 2304, 768, 0, 0);
        k_attn<<<dim3(16, 12, 2), 64, 0, stream>>>(qkv, hb);
        k_gemm<<<dim3(6, 16), 256, 0, stream>>>(hb, p[4], p[5], x, x,
                                                2048, 768, 768, 0, 0);
        k_layernorm<<<2048, 256, 0, stream>>>(x, p[6], p[7], ln);
        k_gemm<<<dim3(24, 16), 256, 0, stream>>>(ln, p[8], p[9], nullptr, hb,
                                                 2048, 3072, 768, 0, 1);
        k_gemm<<<dim3(6, 16), 256, 0, stream>>>(hb, p[10], p[11], x, x,
                                                2048, 768, 3072, 0, 0);
    }

    k_layernorm<<<2048, 256, 0, stream>>>(x, lnf_g, lnf_b, ln);
    // logits = ln @ wte^T  (wte is [V][K] -> blayout 1), N=50257 tail guarded
    k_gemm<<<dim3(393, 16), 256, 0, stream>>>(ln, wte, nullptr, nullptr, logits,
                                              2048, 50257, 768, 1, 0);
    k_rowloss<<<2048, 256, 0, stream>>>(logits, tgt, rl);
    k_finloss<<<1, 256, 0, stream>>>(rl, logits + (size_t)2048 * 50257);
}
